// LigerEAFTFusedLinearCE_16758962389546
// MI455X (gfx1250) — matrix operand order repaired and checked by reading, not compiled, not run on hardware
//
#include <hip/hip_runtime.h>

#define T_TOK 4096
#define H_DIM 1024
#define V_VOC 50257
#define TOPK_K 20
#define LDS_STRIDE 1032   // u16 per row: 1024 + 8 pad -> 2064B (16B aligned, 4-dword bank skew)
#define L2E 1.4426950408889634f

typedef unsigned short u16;
typedef __bf16 v16bf  __attribute__((ext_vector_type(16)));
typedef float  v8f    __attribute__((ext_vector_type(8)));
typedef u16    u16x8  __attribute__((ext_vector_type(8)));
typedef u16    u16x16 __attribute__((ext_vector_type(16)));

static __device__ __forceinline__ u16 f2bf(float f) {
  unsigned u = __float_as_uint(f);
  u += 0x7FFFu + ((u >> 16) & 1u);           // round-to-nearest-even to bf16
  return (u16)(u >> 16);
}
static __device__ __forceinline__ float bf2f(u16 h) {
  return __uint_as_float(((unsigned)h) << 16);
}
static __device__ __forceinline__ v16bf pack16(u16x8 a, u16x8 b) {
  u16x16 v = __builtin_shufflevector(a, b, 0,1,2,3,4,5,6,7,8,9,10,11,12,13,14,15);
  return __builtin_bit_cast(v16bf, v);
}

// ---------------------------------------------------------------------------
// Phase 0: one-time split of W (fp32) into bf16 hi/lo planes; zero accumulators
// ---------------------------------------------------------------------------
__global__ __launch_bounds__(256) void eaft_convert_w(const float* __restrict__ W,
                                                      u16* __restrict__ Whi,
                                                      u16* __restrict__ Wlo,
                                                      float* __restrict__ accum) {
  if (blockIdx.x == 0 && threadIdx.x == 0) { accum[0] = 0.0f; accum[1] = 0.0f; }
  const size_t n4 = ((size_t)V_VOC * H_DIM) >> 2;
  const size_t stride = (size_t)gridDim.x * blockDim.x;
  for (size_t i = (size_t)blockIdx.x * blockDim.x + threadIdx.x; i < n4; i += stride) {
    const float4 v = ((const float4*)W)[i];
    u16 h0 = f2bf(v.x), h1 = f2bf(v.y), h2 = f2bf(v.z), h3 = f2bf(v.w);
    ushort4 hi = make_ushort4(h0, h1, h2, h3);
    ushort4 lo = make_ushort4(f2bf(v.x - bf2f(h0)), f2bf(v.y - bf2f(h1)),
                              f2bf(v.z - bf2f(h2)), f2bf(v.w - bf2f(h3)));
    ((ushort4*)Whi)[i] = hi;
    ((ushort4*)Wlo)[i] = lo;
  }
}

// ---------------------------------------------------------------------------
// Phase 1: per-token target logit  tgt[t] = dot(x[t], W[y[t]])  (fp32)
// One wave per token.
// ---------------------------------------------------------------------------
__global__ __launch_bounds__(256) void eaft_target(const float* __restrict__ x,
                                                   const int* __restrict__ y,
                                                   const float* __restrict__ W,
                                                   float* __restrict__ tgt) {
  const int wave = threadIdx.x >> 5, lane = threadIdx.x & 31;
  const int t = blockIdx.x * 8 + wave;
  int yv = y[t];
  if (yv < 0 || yv >= V_VOC) yv = 0;
  const float4* xr = (const float4*)(x + (size_t)t * H_DIM);
  const float4* wr = (const float4*)(W + (size_t)yv * H_DIM);
  float s = 0.f;
  for (int k = lane; k < (H_DIM / 4); k += 32) {
    float4 a = xr[k], b = wr[k];
    s = fmaf(a.x, b.x, fmaf(a.y, b.y, fmaf(a.z, b.z, fmaf(a.w, b.w, s))));
  }
  for (int off = 16; off; off >>= 1) s += __shfl_down(s, off);
  if (lane == 0) tgt[t] = s;
}

// ---------------------------------------------------------------------------
// Phase 2: streaming split-bf16 WMMA GEMM with fused online softmax,
// count-above-target (top-k test) and EAFT-weighted CE accumulation.
// Block = 32 tokens, 8 waves; wave owns a 16-column slice of V per tile step.
// ---------------------------------------------------------------------------
__global__ __launch_bounds__(256) void eaft_main(const float* __restrict__ x,
                                                 const int* __restrict__ y,
                                                 const u16* __restrict__ Whi,
                                                 const u16* __restrict__ Wlo,
                                                 const float* __restrict__ tgt,
                                                 float* __restrict__ accum) {
  extern __shared__ char smem[];
  u16* sXhi = (u16*)smem;                              // 32 x LDS_STRIDE
  u16* sXlo = sXhi + 32 * LDS_STRIDE;
  float* sStat = (float*)(smem + (size_t)2 * 32 * LDS_STRIDE * sizeof(u16)); // [8][32][3]

  const int t0 = blockIdx.x * 32;
  const int tid = threadIdx.x;

  // Stage + split x tile (32 x 1024 fp32 -> bf16 hi/lo in LDS)
  for (int i = tid; i < 32 * H_DIM; i += 256) {
    int r = i >> 10, c = i & (H_DIM - 1);
    float v = x[(size_t)(t0 + r) * H_DIM + c];
    u16 hi = f2bf(v);
    sXhi[r * LDS_STRIDE + c] = hi;
    sXlo[r * LDS_STRIDE + c] = f2bf(v - bf2f(hi));
  }
  __syncthreads();

  const int wave = tid >> 5, lane = tid & 31;
  const int col0 = lane & 15;
  const int hiHalf = lane >> 4;
  const int arow = lane & 15;
  const int kA = hiHalf ? 8 : 0;     // A lanes 16..31 carry K {8..15, 24..31}
  const int kB = hiHalf ? 8 : 0;     // B lanes 16..31 carry K {8..15, 24..31}
  const int rbase = hiHalf ? 8 : 0;  // C row mapping: vgpr r -> token rbase+r

  float tgtA[8], tgtB[8];
  float mxA[8], smA[8], cnA[8], mxB[8], smB[8], cnB[8];
#pragma unroll
  for (int r = 0; r < 8; ++r) {
    tgtA[r] = tgt[t0 + rbase + r];
    tgtB[r] = tgt[t0 + 16 + rbase + r];
    mxA[r] = -__builtin_inff(); smA[r] = 0.f; cnA[r] = 0.f;
    mxB[r] = -__builtin_inff(); smB[r] = 0.f; cnB[r] = 0.f;
  }

  const int NT = (V_VOC + 15) / 16;
  for (int j = wave; j < NT; j += 8) {
    const int vb = j * 16;
    int col = vb + col0;
    const bool colok = col < V_VOC;
    if (!colok) col = V_VOC - 1;
    const u16* bh = Whi + (size_t)col * H_DIM + kB;
    const u16* bl = Wlo + (size_t)col * H_DIM + kB;
    {
      int jn = j + 8;
      if (jn < NT) {
        int cn = jn * 16 + col0; if (cn >= V_VOC) cn = V_VOC - 1;
        __builtin_prefetch(Whi + (size_t)cn * H_DIM + kB, 0, 3);
        __builtin_prefetch(Wlo + (size_t)cn * H_DIM + kB, 0, 3);
      }
    }
    v8f cAcc = {};
    v8f cAcc2 = {};
    const u16* a0h = sXhi + arow * LDS_STRIDE + kA;
    const u16* a0l = sXlo + arow * LDS_STRIDE + kA;
    const u16* a1h = sXhi + (16 + arow) * LDS_STRIDE + kA;
    const u16* a1l = sXlo + (16 + arow) * LDS_STRIDE + kA;
#pragma unroll 4
    for (int k0 = 0; k0 < H_DIM; k0 += 32) {
      v16bf Bh  = pack16(*(const u16x8*)(bh  + k0), *(const u16x8*)(bh  + k0 + 16));
      v16bf Bl  = pack16(*(const u16x8*)(bl  + k0), *(const u16x8*)(bl  + k0 + 16));
      v16bf A0h = pack16(*(const u16x8*)(a0h + k0), *(const u16x8*)(a0h + k0 + 16));
      v16bf A0l = pack16(*(const u16x8*)(a0l + k0), *(const u16x8*)(a0l + k0 + 16));
      v16bf A1h = pack16(*(const u16x8*)(a1h + k0), *(const u16x8*)(a1h + k0 + 16));
      v16bf A1l = pack16(*(const u16x8*)(a1l + k0), *(const u16x8*)(a1l + k0 + 16));
      // fp32-ish product via 3-term bf16 split: hi*hi + hi*lo + lo*hi
      cAcc  = __builtin_amdgcn_wmma_f32_16x16x32_bf16(false, A0h, false, Bh, (short)0, cAcc,  false, false);
      cAcc  = __builtin_amdgcn_wmma_f32_16x16x32_bf16(false, A0h, false, Bl, (short)0, cAcc,  false, false);
      cAcc  = __builtin_amdgcn_wmma_f32_16x16x32_bf16(false, A0l, false, Bh, (short)0, cAcc,  false, false);
      cAcc2 = __builtin_amdgcn_wmma_f32_16x16x32_bf16(false, A1h, false, Bh, (short)0, cAcc2, false, false);
      cAcc2 = __builtin_amdgcn_wmma_f32_16x16x32_bf16(false, A1h, false, Bl, (short)0, cAcc2, false, false);
      cAcc2 = __builtin_amdgcn_wmma_f32_16x16x32_bf16(false, A1l, false, Bh, (short)0, cAcc2, false, false);
    }
    if (colok) {
#pragma unroll
      for (int r = 0; r < 8; ++r) {
        float v = cAcc[r];
        cnA[r] += (v > tgtA[r]) ? 1.0f : 0.0f;
        float mm = fmaxf(mxA[r], v);
        smA[r] = smA[r] * exp2f((mxA[r] - mm) * L2E) + exp2f((v - mm) * L2E);
        mxA[r] = mm;
        float v2 = cAcc2[r];
        cnB[r] += (v2 > tgtB[r]) ? 1.0f : 0.0f;
        float mm2 = fmaxf(mxB[r], v2);
        smB[r] = smB[r] * exp2f((mxB[r] - mm2) * L2E) + exp2f((v2 - mm2) * L2E);
        mxB[r] = mm2;
      }
    }
  }

  // Merge across the 16 lanes of each N-group (xor masks stay inside halves)
#pragma unroll
  for (int r = 0; r < 8; ++r) {
    for (int m = 1; m <= 8; m <<= 1) {
      float om = __shfl_xor(mxA[r], m), os = __shfl_xor(smA[r], m), oc = __shfl_xor(cnA[r], m);
      float mm = fmaxf(mxA[r], om);
      smA[r] = smA[r] * exp2f((mxA[r] - mm) * L2E) + os * exp2f((om - mm) * L2E);
      mxA[r] = mm; cnA[r] += oc;
      om = __shfl_xor(mxB[r], m); os = __shfl_xor(smB[r], m); oc = __shfl_xor(cnB[r], m);
      mm = fmaxf(mxB[r], om);
      smB[r] = smB[r] * exp2f((mxB[r] - mm) * L2E) + os * exp2f((om - mm) * L2E);
      mxB[r] = mm; cnB[r] += oc;
    }
  }
  if (col0 == 0) {
#pragma unroll
    for (int r = 0; r < 8; ++r) {
      int ta = rbase + r, tb = 16 + rbase + r;
      sStat[(wave * 32 + ta) * 3 + 0] = mxA[r];
      sStat[(wave * 32 + ta) * 3 + 1] = smA[r];
      sStat[(wave * 32 + ta) * 3 + 2] = cnA[r];
      sStat[(wave * 32 + tb) * 3 + 0] = mxB[r];
      sStat[(wave * 32 + tb) * 3 + 1] = smB[r];
      sStat[(wave * 32 + tb) * 3 + 2] = cnB[r];
    }
  }
  __syncthreads();

  // Wave 0 merges across the 8 waves, then computes the weighted CE terms
  if (tid < 32) {
    float m = -__builtin_inff(), s = 0.f, c = 0.f;
    for (int w = 0; w < 8; ++w) {
      float om = sStat[(w * 32 + tid) * 3 + 0];
      float os = sStat[(w * 32 + tid) * 3 + 1];
      float oc = sStat[(w * 32 + tid) * 3 + 2];
      float mm = fmaxf(m, om);
      s = s * exp2f((m - mm) * L2E) + os * exp2f((om - mm) * L2E);
      m = mm; c += oc;
    }
    int yv = y[t0 + tid];
    bool valid = (yv != -100);
    float tg = tgt[t0 + tid];
    float lse = m + log2f(s) * (1.0f / L2E);
    float logpt = tg - lse;
    float p = exp2f(logpt * L2E);                       // p_target; ALPHA == 1
    bool intop = c < ((float)TOPK_K - 0.5f);            // #{logits > tgt} < k
    float w = (valid && intop) ? p : 0.0f;
    float contrib = w * (-logpt);
    float vc = valid ? 1.0f : 0.0f;
    for (int off = 16; off; off >>= 1) {
      contrib += __shfl_down(contrib, off);
      vc += __shfl_down(vc, off);
    }
    if (tid == 0) { atomicAdd(&accum[0], contrib); atomicAdd(&accum[1], vc); }
  }
}

// ---------------------------------------------------------------------------
// Phase 3: final mean
// ---------------------------------------------------------------------------
__global__ void eaft_finalize(const float* __restrict__ accum, float* __restrict__ out) {
  out[0] = accum[0] / fmaxf(accum[1], 1.0f);
}

extern "C" void kernel_launch(void* const* d_in, const int* in_sizes, int n_in,
                              void* d_out, int out_size, void* d_ws, size_t ws_size,
                              hipStream_t stream) {
  (void)in_sizes; (void)n_in; (void)out_size; (void)ws_size;
  const float* x = (const float*)d_in[0];
  const int*   y = (const int*)d_in[1];
  const float* W = (const float*)d_in[2];
  float* out = (float*)d_out;

  const size_t VH = (size_t)V_VOC * H_DIM;
  u16* Whi = (u16*)d_ws;
  u16* Wlo = Whi + VH;
  float* tgt = (float*)(Wlo + VH);
  float* accum = tgt + T_TOK;

  eaft_convert_w<<<2048, 256, 0, stream>>>(W, Whi, Wlo, accum);
  eaft_target<<<T_TOK / 8, 256, 0, stream>>>(x, y, W, tgt);
  const size_t smem_bytes = (size_t)2 * 32 * LDS_STRIDE * sizeof(u16)
                          + (size_t)8 * 32 * 3 * sizeof(float);   // 135,168 B
  eaft_main<<<T_TOK / 32, 256, smem_bytes, stream>>>(x, y, Whi, Wlo, tgt, accum);
  eaft_finalize<<<1, 1, 0, stream>>>(accum, out);
}
